// SpatialModel_52261162058269
// MI455X (gfx1250) — matrix operand order repaired and checked by reading, not compiled, not run on hardware
//
#include <hip/hip_runtime.h>
#include <hip/hip_bf16.h>

// ===== PCT encoder forward for MI455X (gfx1250, wave32, WMMA) =====
// All heavy channel-mixing ops go through v_wmma_f32_16x16x32_f16.
// f32 tensors in HBM, f16 staged in LDS with conflict-free padded layouts;
// fragments loaded as 16B ds_load_b128 pairs. BN (training mode, global
// stats): GEMM+stat-atomics -> finalize(scale,bias) -> fused into next
// GEMM's operand load. Dead branches (sa3, sa4, new_feature_return_*) skipped.

typedef __attribute__((ext_vector_type(16))) _Float16 v16h;
typedef __attribute__((ext_vector_type(8)))  _Float16 v8h;
typedef __attribute__((ext_vector_type(2)))  _Float16 v2h;
typedef __attribute__((ext_vector_type(8)))  float    v8f;

#define NPTS   8192
#define NBATCH 64
#define BNCOLS (NBATCH * NPTS)   // 524288

#define LDA_G 40   // padded row (halves) for 32-K tiles: 20-dword stride, bank-conflict-free
#define LDA_C 72   // padded row (halves) for 64-K tiles: 36-dword stride, bank-conflict-free

__device__ __forceinline__ float relu_f(float v) { return v > 0.f ? v : 0.f; }

// ---- WMMA fragment helpers (layouts per CDNA5 ISA 7.12.2, wave32) ----
// A (16x32 f16): lanes 0-15: M=lane, halves 0..7 = K kb+0..7, 8..15 = K kb+16..23
//                lanes16-31: M=lane-16, K offset by +8 / +24
__device__ __forceinline__ v16h load_a_frag(const _Float16* a, int lda, int m0, int k0, int lane) {
  int lh = lane & 15, hi = (lane >> 4) & 1;
  const _Float16* row = a + (m0 + lh) * lda;
  int kb0 = k0 + (hi ? 8 : 0);
  int kb1 = k0 + (hi ? 24 : 16);
  v8h lo = *(const v8h*)__builtin_assume_aligned(row + kb0, 16);
  v8h hi8 = *(const v8h*)__builtin_assume_aligned(row + kb1, 16);
  return __builtin_shufflevector(lo, hi8, 0,1,2,3,4,5,6,7,8,9,10,11,12,13,14,15);
}
// B (32x16 f16) from TRANSPOSED tile bt[n][k]: halves 0..7 = K kb+0..7, 8..15 = K kb+16..23 (hi lanes +8)
__device__ __forceinline__ v16h load_b_fragT(const _Float16* bt, int ldb, int n0, int k0, int lane) {
  int lh = lane & 15, hi = (lane >> 4) & 1;
  const _Float16* row = bt + (n0 + lh) * ldb + k0 + (hi ? 8 : 0);
  v8h lo = *(const v8h*)__builtin_assume_aligned(row, 16);
  v8h hi8 = *(const v8h*)__builtin_assume_aligned(row + 16, 16);
  return __builtin_shufflevector(lo, hi8, 0,1,2,3,4,5,6,7,8,9,10,11,12,13,14,15);
}
__device__ __forceinline__ v8f wmma16(v16h a, v16h b, v8f c) {
  return __builtin_amdgcn_wmma_f32_16x16x32_f16(false, a, false, b, (short)0, c, false, false);
}

// ---- generic WMMA GEMM: Y[M,NC] = W[M,K] @ g(X[K,NC]) (+outBias) (+BN stats)
// g(x) = relu(inScale*x+inBias) if inScale, optionally relu only, else identity.
__global__ __launch_bounds__(256)
void pct_gemm(const float* __restrict__ W, const float* __restrict__ X,
              float* __restrict__ Y,
              const float* __restrict__ inScale, const float* __restrict__ inBias, int inRelu,
              const float* __restrict__ outBias, float* __restrict__ statsPtr,
              int M, int K, int NC) {
  __shared__ __attribute__((aligned(16))) _Float16 aT[64 * LDA_G];
  __shared__ __attribute__((aligned(16))) _Float16 bTt[32 * LDA_G];
  __shared__ float ssum[64], ssq[64];
  const int tid  = threadIdx.x;
  const int lane = tid & 31, wave = tid >> 5;
  const int row0 = blockIdx.y * 64;
  const int col0 = blockIdx.x * 32;
  const int m0   = (wave & 3) * 16;
  const int n0   = (wave >> 2) * 16;

  v8f acc = {};
  for (int k0 = 0; k0 < K; k0 += 32) {
#pragma unroll
    for (int t = 0; t < 4; ++t) {            // stage W block 64x32 as K-pairs (float2 -> b32)
      int p = tid + t * 256;                 // 1024 pairs
      int m = p >> 4, kk = (p & 15) * 2;
      float2 wv = *(const float2*)&W[(size_t)(row0 + m) * K + k0 + kk];
      *(v2h*)&aT[m * LDA_G + kk] = (v2h){(_Float16)wv.x, (_Float16)wv.y};
    }
#pragma unroll
    for (int t = 0; t < 2; ++t) {            // stage X block 32x32 (transposed), fused BN+ReLU
      int p = tid + t * 256;                 // 512 float2 column-pair loads (coalesced)
      int kk = p >> 4, nn = (p & 15) * 2;
      float2 xv = *(const float2*)&X[(size_t)(k0 + kk) * NC + col0 + nn];
      float v0 = xv.x, v1 = xv.y;
      if (inScale) { float s = inScale[k0 + kk], bb = inBias[k0 + kk];
                     v0 = s * v0 + bb; v1 = s * v1 + bb; }
      if (inRelu)  { v0 = relu_f(v0); v1 = relu_f(v1); }
      bTt[nn * LDA_G + kk]       = (_Float16)v0;
      bTt[(nn + 1) * LDA_G + kk] = (_Float16)v1;
      if (k0 + 32 < K)                       // gfx1250 global_prefetch of next K tile
        __builtin_prefetch(&X[(size_t)(k0 + 32 + kk) * NC + col0 + nn], 0, 1);
    }
    __syncthreads();
    v16h af = load_a_frag(aT, LDA_G, m0, 0, lane);
    v16h bf = load_b_fragT(bTt, LDA_G, n0, 0, lane);
    acc = wmma16(af, bf, acc);
    __syncthreads();
  }
  if (tid < 64) { ssum[tid] = 0.f; ssq[tid] = 0.f; }
  __syncthreads();
  int lh = lane & 15, hi = (lane >> 4) & 1;
  int colg = col0 + n0 + lh;
#pragma unroll
  for (int r = 0; r < 8; ++r) {              // C/D layout: vgpr r -> row m0+r (+8 for hi lanes)
    int chl = m0 + (hi ? 8 : 0) + r;
    int chg = row0 + chl;
    float v = acc[r];
    if (outBias) v += outBias[chg];
    Y[(size_t)chg * NC + colg] = v;
    if (statsPtr) { atomicAdd(&ssum[chl], v); atomicAdd(&ssq[chl], v * v); }
  }
  __syncthreads();
  if (statsPtr && tid < 64) {
    atomicAdd(&statsPtr[row0 + tid],     ssum[tid]);
    atomicAdd(&statsPtr[M + row0 + tid], ssq[tid]);
  }
}

// ---- fused conv1(3->64)+BN1+ReLU+conv2(64->64) with BN2 stat accumulation
__global__ __launch_bounds__(256)
void pct_conv12(const float* __restrict__ x, const float* __restrict__ W1,
                const float* __restrict__ s1, const float* __restrict__ bb1,
                const float* __restrict__ W2, float* __restrict__ y2,
                float* __restrict__ statsPtr) {
  __shared__ __attribute__((aligned(16))) _Float16 aT[64 * LDA_C];
  __shared__ __attribute__((aligned(16))) _Float16 bTt[32 * LDA_C];
  __shared__ float xs[3][32];
  __shared__ float w1s[64 * 3], s1s[64], b1s[64];
  __shared__ float ssum[64], ssq[64];
  const int tid = threadIdx.x, lane = tid & 31, wave = tid >> 5;
  const int col0 = blockIdx.x * 32;
  const int b = col0 / NPTS;
  const int n0 = col0 - b * NPTS;

  if (tid < 192) w1s[tid] = W1[tid];
  if (tid < 64) { s1s[tid] = s1[tid]; b1s[tid] = bb1[tid]; ssum[tid] = 0.f; ssq[tid] = 0.f; }
  if (tid < 96) { int j = tid >> 5, nn = tid & 31;
    xs[j][nn] = x[(size_t)b * 3 * NPTS + (size_t)j * NPTS + n0 + nn]; }
#pragma unroll
  for (int t = 0; t < 8; ++t) {              // stage W2 64x64 as K-pairs
    int p = tid + t * 256;                   // 2048 pairs
    int m = p >> 5, kk = (p & 31) * 2;
    float2 wv = *(const float2*)&W2[m * 64 + kk];
    *(v2h*)&aT[m * LDA_C + kk] = (v2h){(_Float16)wv.x, (_Float16)wv.y};
  }
  __syncthreads();
#pragma unroll
  for (int t = 0; t < 4; ++t) {              // t1 = relu(bn1(W1@x)) on the fly -> bTt[n][k]
    int p = tid + t * 256;                   // 1024 K-pairs
    int nn = p >> 5, kk = (p & 31) * 2;
    float u0 = w1s[kk*3]     * xs[0][nn] + w1s[kk*3+1]     * xs[1][nn] + w1s[kk*3+2]     * xs[2][nn];
    float u1 = w1s[kk*3+3]   * xs[0][nn] + w1s[kk*3+4]     * xs[1][nn] + w1s[kk*3+5]     * xs[2][nn];
    u0 = relu_f(s1s[kk] * u0 + b1s[kk]);
    u1 = relu_f(s1s[kk+1] * u1 + b1s[kk+1]);
    *(v2h*)&bTt[nn * LDA_C + kk] = (v2h){(_Float16)u0, (_Float16)u1};
  }
  __syncthreads();
  const int m0 = (wave & 3) * 16, n0w = (wave >> 2) * 16;
  v8f acc = {};
#pragma unroll
  for (int k0 = 0; k0 < 64; k0 += 32) {
    v16h af = load_a_frag(aT, LDA_C, m0, k0, lane);
    v16h bf = load_b_fragT(bTt, LDA_C, n0w, k0, lane);
    acc = wmma16(af, bf, acc);
  }
  int lh = lane & 15, hi = (lane >> 4) & 1;
  int colg = col0 + n0w + lh;
#pragma unroll
  for (int r = 0; r < 8; ++r) {
    int chl = m0 + (hi ? 8 : 0) + r;
    float v = acc[r];
    y2[(size_t)chl * BNCOLS + colg] = v;
    atomicAdd(&ssum[chl], v); atomicAdd(&ssq[chl], v * v);
  }
  __syncthreads();
  if (tid < 64) { atomicAdd(&statsPtr[tid], ssum[tid]); atomicAdd(&statsPtr[64 + tid], ssq[tid]); }
}

// ---- x moments (BN1 is a linear map of x; stats from 3x3 second moment) ----
__global__ __launch_bounds__(256)
void pct_xmom(const float* __restrict__ x, float* __restrict__ stats) {
  __shared__ float acc[12];
  int tid = threadIdx.x;
  if (tid < 12) acc[tid] = 0.f;
  __syncthreads();
  float s0=0,s1=0,s2=0,p00=0,p01=0,p02=0,p11=0,p12=0,p22=0;
  for (int col = blockIdx.x * 256 + tid; col < BNCOLS; col += gridDim.x * 256) {
    int b = col / NPTS, n = col % NPTS;
    size_t xb = (size_t)b * 3 * NPTS + n;
    float x0 = x[xb], x1 = x[xb + NPTS], x2 = x[xb + 2 * NPTS];
    s0+=x0; s1+=x1; s2+=x2;
    p00+=x0*x0; p01+=x0*x1; p02+=x0*x2; p11+=x1*x1; p12+=x1*x2; p22+=x2*x2;
  }
  atomicAdd(&acc[0],s0); atomicAdd(&acc[1],s1); atomicAdd(&acc[2],s2);
  atomicAdd(&acc[3],p00); atomicAdd(&acc[4],p01); atomicAdd(&acc[5],p02);
  atomicAdd(&acc[6],p11); atomicAdd(&acc[7],p12); atomicAdd(&acc[8],p22);
  __syncthreads();
  if (tid < 9) atomicAdd(&stats[tid], acc[tid]);
}

__global__ void pct_bn1fin(const float* __restrict__ stats, const float* __restrict__ W1,
                           const float* __restrict__ g, const float* __restrict__ be,
                           float* __restrict__ sb) {
  int c = threadIdx.x;
  if (c >= 64) return;
  const float inv = 1.f / (float)BNCOLS;
  float mx0=stats[0]*inv, mx1=stats[1]*inv, mx2=stats[2]*inv;
  float M00=stats[3]*inv, M01=stats[4]*inv, M02=stats[5]*inv;
  float M11=stats[6]*inv, M12=stats[7]*inv, M22=stats[8]*inv;
  float w0=W1[c*3], w1=W1[c*3+1], w2=W1[c*3+2];
  float mean = w0*mx0 + w1*mx1 + w2*mx2;
  float e2 = w0*w0*M00 + w1*w1*M11 + w2*w2*M22 + 2.f*(w0*w1*M01 + w0*w2*M02 + w1*w2*M12);
  float var = e2 - mean * mean;
  float sc = g[c] * rsqrtf(var + 1e-5f);
  sb[c] = sc; sb[64 + c] = be[c] - mean * sc;
}

__global__ void pct_bnfin(const float* __restrict__ stats, const float* __restrict__ g,
                          const float* __restrict__ be, float* __restrict__ sb,
                          int C, float count) {
  int c = blockIdx.x * blockDim.x + threadIdx.x;
  if (c >= C) return;
  float mean = stats[c] / count;
  float var  = stats[C + c] / count - mean * mean;
  float sc = g[c] * rsqrtf(var + 1e-5f);
  sb[c] = sc; sb[C + c] = be[c] - mean * sc;
}

// ---- grouping 1: knn top-8 over 8192, build feat [128][B*64*8] ----
__global__ __launch_bounds__(256)
void pct_group1(const float* __restrict__ x, const int* __restrict__ idx1,
                const float* __restrict__ y2, const float* __restrict__ s2,
                const float* __restrict__ bb2,
                float* __restrict__ nx1, float* __restrict__ ft1) {
  const int tid = threadIdx.x;
  const int bs = blockIdx.x;           // b*64+s
  const int b = bs >> 6;
  __shared__ float ctr[64];
  __shared__ float cd[2048]; __shared__ int cix[2048];
  __shared__ float rv[256];  __shared__ int rp[256];
  __shared__ int sel[8];
  const int ci = idx1[bs];
  const size_t xb = (size_t)b * 3 * NPTS;
  const float c0 = x[xb + ci], c1 = x[xb + NPTS + ci], c2 = x[xb + 2 * NPTS + ci];
  if (tid < 3) nx1[bs * 3 + tid] = (tid == 0 ? c0 : (tid == 1 ? c1 : c2));
  if (tid < 64)
    ctr[tid] = relu_f(s2[tid] * y2[(size_t)tid * BNCOLS + (size_t)b * NPTS + ci] + bb2[tid]);

  float bd[8]; int bix[8];
#pragma unroll
  for (int j = 0; j < 8; ++j) { bd[j] = 3.0e38f; bix[j] = 0; }
  for (int m = tid; m < NPTS; m += 256) {
    float dx = x[xb + m] - c0, dy = x[xb + NPTS + m] - c1, dz = x[xb + 2 * NPTS + m] - c2;
    float d2 = dx*dx + dy*dy + dz*dz;
    if (d2 < bd[7]) {
      bd[7] = d2; bix[7] = m;
      for (int j = 7; j > 0; --j) {
        if (bd[j] < bd[j-1]) { float tv=bd[j]; bd[j]=bd[j-1]; bd[j-1]=tv;
                               int ti=bix[j]; bix[j]=bix[j-1]; bix[j-1]=ti; }
        else break;
      }
    }
  }
#pragma unroll
  for (int j = 0; j < 8; ++j) { cd[tid*8 + j] = bd[j]; cix[tid*8 + j] = bix[j]; }
  __syncthreads();
  for (int r = 0; r < 8; ++r) {          // 8 rounds of parallel argmin over 2048
    float v = cd[tid*8]; int p = tid*8;
#pragma unroll
    for (int j = 1; j < 8; ++j) if (cd[tid*8+j] < v) { v = cd[tid*8+j]; p = tid*8+j; }
    rv[tid] = v; rp[tid] = p;
    __syncthreads();
    for (int st = 128; st > 0; st >>= 1) {
      if (tid < st && rv[tid+st] < rv[tid]) { rv[tid]=rv[tid+st]; rp[tid]=rp[tid+st]; }
      __syncthreads();
    }
    if (tid == 0) { sel[r] = cix[rp[0]]; cd[rp[0]] = 3.0e38f; }
    __syncthreads();
  }
  const int colb = bs * 8;
  const int NC1 = NBATCH * 64 * 8;
  for (int t = tid; t < 512; t += 256) {
    int k = t >> 6, c = t & 63;
    int pi = sel[k];
    float pv = relu_f(s2[c] * y2[(size_t)c * BNCOLS + (size_t)b * NPTS + pi] + bb2[c]);
    ft1[(size_t)c * NC1 + colb + k]        = pv - ctr[c];   // gn channels 0..63
    ft1[(size_t)(64 + c) * NC1 + colb + k] = ctr[c];        // ctr channels 64..127
  }
}

// ---- grouping 2: knn top-8 over 64 samples, feat [256][B*32*8] ----
__global__ __launch_bounds__(64)
void pct_group2(const float* __restrict__ nx1, const int* __restrict__ idx2,
                const float* __restrict__ f0, float* __restrict__ nx2,
                float* __restrict__ ft2) {
  const int tid = threadIdx.x;
  const int bs = blockIdx.x;            // b*32+s
  const int b = bs >> 5;
  __shared__ float ctr[128];
  __shared__ float cd[64]; __shared__ float rv[64]; __shared__ int rp[64];
  __shared__ int sel[8];
  const int ci = idx2[bs];
  const float c0 = nx1[(b*64+ci)*3], c1 = nx1[(b*64+ci)*3+1], c2 = nx1[(b*64+ci)*3+2];
  if (tid < 3) nx2[bs * 3 + tid] = (tid == 0 ? c0 : (tid == 1 ? c1 : c2));
  const int NCF0 = NBATCH * 64;
  ctr[tid]      = f0[(size_t)tid * NCF0 + b * 64 + ci];
  ctr[tid + 64] = f0[(size_t)(tid + 64) * NCF0 + b * 64 + ci];
  {
    float dx = nx1[(b*64+tid)*3] - c0, dy = nx1[(b*64+tid)*3+1] - c1, dz = nx1[(b*64+tid)*3+2] - c2;
    cd[tid] = dx*dx + dy*dy + dz*dz;
  }
  __syncthreads();
  for (int r = 0; r < 8; ++r) {
    rv[tid] = cd[tid]; rp[tid] = tid;
    __syncthreads();
    for (int st = 32; st > 0; st >>= 1) {
      if (tid < st && rv[tid+st] < rv[tid]) { rv[tid]=rv[tid+st]; rp[tid]=rp[tid+st]; }
      __syncthreads();
    }
    if (tid == 0) { sel[r] = rp[0]; cd[rp[0]] = 3.0e38f; }
    __syncthreads();
  }
  const int colb = bs * 8;
  const int NC2 = NBATCH * 32 * 8;
  for (int t = tid; t < 1024; t += 64) {
    int k = t >> 7, c = t & 127;
    float pv = f0[(size_t)c * NCF0 + b * 64 + sel[k]];
    ft2[(size_t)c * NC2 + colb + k]         = pv - ctr[c];
    ft2[(size_t)(128 + c) * NC2 + colb + k] = ctr[c];
  }
}

// ---- apply BN+ReLU and max over 8 neighbors ----
__global__ void pct_maxpool(const float* __restrict__ Yin, const float* __restrict__ sc,
                            const float* __restrict__ bi, float* __restrict__ out,
                            int M, int groups) {
  int idx = blockIdx.x * blockDim.x + threadIdx.x;
  if (idx >= M * groups) return;
  int c = idx / groups, g = idx % groups;
  const float s = sc[c], b = bi[c];
  size_t base = (size_t)c * groups * 8 + (size_t)g * 8;
  float m = -3.0e38f;
#pragma unroll
  for (int k = 0; k < 8; ++k) m = fmaxf(m, relu_f(s * Yin[base + k] + b));
  out[(size_t)c * groups + g] = m;
}

__global__ void pct_posemb(const float* __restrict__ nx2, const float* __restrict__ wpos,
                           const float* __restrict__ bpos, float* __restrict__ emb) {
  int idx = blockIdx.x * blockDim.x + threadIdx.x;
  const int NC = NBATCH * 32;
  if (idx >= 256 * NC) return;
  int c = idx / NC, col = idx % NC;
  emb[idx] = wpos[c*3] * nx2[col*3] + wpos[c*3+1] * nx2[col*3+1] +
             wpos[c*3+2] * nx2[col*3+2] + bpos[c];
}

__global__ void pct_addh(const float* __restrict__ a, const float* __restrict__ sc,
                         const float* __restrict__ bi, const float* __restrict__ e,
                         float* __restrict__ out, int NC) {
  int idx = blockIdx.x * blockDim.x + threadIdx.x;
  if (idx >= 256 * NC) return;
  int c = idx / NC;
  float v = a[idx];
  if (sc) v = relu_f(sc[c] * v + bi[c]);
  out[idx] = v + e[idx];
}

// ---- offset attention core: e=K^T K, softmax, column-norm, hx = h - V a ----
__global__ __launch_bounds__(256)
void pct_attn(const float* __restrict__ qk, const float* __restrict__ v,
              const float* __restrict__ h, float* __restrict__ hx) {
  __shared__ float qs[64 * 32];
  __shared__ float vs[256 * 32];
  __shared__ float e[32 * 32];
  const int tid = threadIdx.x;
  const int b = blockIdx.x;
  const int NC = NBATCH * 32;
#pragma unroll
  for (int t = 0; t < 8; ++t) {
    int idx = tid + t * 256; int c = idx >> 5, n = idx & 31;
    qs[idx] = qk[(size_t)c * NC + b * 32 + n];
  }
#pragma unroll
  for (int t = 0; t < 32; ++t) {
    int idx = tid + t * 256; int c = idx >> 5, n = idx & 31;
    vs[idx] = v[(size_t)c * NC + b * 32 + n];
  }
  __syncthreads();
#pragma unroll
  for (int t = 0; t < 4; ++t) {            // e[n][m] = sum_c qk[c][n]*qk[c][m]
    int idx = tid + t * 256; int n = idx >> 5, m = idx & 31;
    float s = 0.f;
    for (int c = 0; c < 64; ++c) s += qs[c * 32 + n] * qs[c * 32 + m];
    e[idx] = s;
  }
  __syncthreads();
  if (tid < 32) {                           // softmax over m (rows)
    int n = tid;
    float mx = -3.0e38f;
    for (int m = 0; m < 32; ++m) mx = fmaxf(mx, e[n*32+m]);
    float s = 0.f;
    for (int m = 0; m < 32; ++m) { float ev = __expf(e[n*32+m] - mx); e[n*32+m] = ev; s += ev; }
    float inv = 1.f / s;
    for (int m = 0; m < 32; ++m) e[n*32+m] *= inv;
  }
  __syncthreads();
  if (tid < 32) {                           // double-normalize: column sums over n
    int m = tid;
    float cs = 0.f;
    for (int n = 0; n < 32; ++n) cs += e[n*32+m];
    float inv = 1.f / (1e-9f + cs);
    for (int n = 0; n < 32; ++n) e[n*32+m] *= inv;
  }
  __syncthreads();
#pragma unroll
  for (int t = 0; t < 32; ++t) {            // hx = h - V a
    int idx = tid + t * 256; int c = idx >> 5, m = idx & 31;
    float xr = 0.f;
    for (int n = 0; n < 32; ++n) xr += vs[c * 32 + n] * e[n * 32 + m];
    hx[(size_t)c * NC + b * 32 + m] = h[(size_t)c * NC + b * 32 + m] - xr;
  }
}

__global__ void pct_saout(const float* __restrict__ h, const float* __restrict__ t,
                          const float* __restrict__ sc, const float* __restrict__ bi,
                          float* __restrict__ out, int NC) {
  int idx = blockIdx.x * blockDim.x + threadIdx.x;
  if (idx >= 256 * NC) return;
  int c = idx / NC;
  out[idx] = h[idx] + relu_f(sc[c] * t[idx] + bi[c]);
}

__global__ void pct_fusemax(const float* __restrict__ y, const float* __restrict__ sc,
                            const float* __restrict__ bi, float* __restrict__ out) {
  int idx = blockIdx.x * blockDim.x + threadIdx.x;   // b*1024 + c
  if (idx >= NBATCH * 1024) return;
  int b = idx >> 10, c = idx & 1023;
  const int NC = NBATCH * 32;
  float s = sc[c], bb = bi[c];
  float m = -3.0e38f;
  for (int n = 0; n < 32; ++n) {
    float v = s * y[(size_t)c * NC + b * 32 + n] + bb;
    v = v > 0.f ? v : 0.2f * v;              // LeakyReLU(0.2)
    m = fmaxf(m, v);
  }
  out[idx] = m;
}

__global__ void pct_emitxpt(const float* __restrict__ xcat, float* __restrict__ out) {
  int idx = blockIdx.x * blockDim.x + threadIdx.x;   // b*512*32 + c*32 + n
  if (idx >= NBATCH * 512 * 32) return;
  int b = idx / (512 * 32);
  int r = idx % (512 * 32);
  int c = r >> 5, n = r & 31;
  out[idx] = xcat[(size_t)c * (NBATCH * 32) + b * 32 + n];
}

__global__ void pct_zero(float* p, int n) {
  int i = blockIdx.x * blockDim.x + threadIdx.x;
  if (i < n) p[i] = 0.f;
}

// ===================== host side =====================
extern "C" void kernel_launch(void* const* d_in, const int* in_sizes, int n_in,
                              void* d_out, int out_size, void* d_ws, size_t ws_size,
                              hipStream_t stream) {
  (void)in_sizes; (void)n_in; (void)out_size; (void)ws_size;
  int i = 0;
  const float* x    = (const float*)d_in[i++];
  const int*   idx1 = (const int*)d_in[i++];
  const int*   idx2 = (const int*)d_in[i++];
  i += 2;  // new_feature_return_1/2 : dead in forward path
  const float* w_c1 = (const float*)d_in[i++];
  const float* g1   = (const float*)d_in[i++];
  const float* b1   = (const float*)d_in[i++];
  const float* w_c2 = (const float*)d_in[i++];
  const float* g2   = (const float*)d_in[i++];
  const float* b2   = (const float*)d_in[i++];
  const float* l0_w1 = (const float*)d_in[i++];
  const float* l0_g1 = (const float*)d_in[i++];
  const float* l0_b1 = (const float*)d_in[i++];
  const float* l0_w2 = (const float*)d_in[i++];
  const float* l0_g2 = (const float*)d_in[i++];
  const float* l0_b2 = (const float*)d_in[i++];
  const float* l1_w1 = (const float*)d_in[i++];
  const float* l1_g1 = (const float*)d_in[i++];
  const float* l1_b1 = (const float*)d_in[i++];
  const float* l1_w2 = (const float*)d_in[i++];
  const float* l1_g2 = (const float*)d_in[i++];
  const float* l1_b2 = (const float*)d_in[i++];
  const float* w_pos = (const float*)d_in[i++];
  const float* b_pos = (const float*)d_in[i++];
  const float* pt_w  = (const float*)d_in[i++];
  const float* pt_g  = (const float*)d_in[i++];
  const float* pt_b  = (const float*)d_in[i++];
  const float* sa1_wqk = (const float*)d_in[i++];
  const float* sa1_wv  = (const float*)d_in[i++];
  const float* sa1_bv  = (const float*)d_in[i++];
  const float* sa1_wt  = (const float*)d_in[i++];
  const float* sa1_bt  = (const float*)d_in[i++];
  const float* sa1_gn  = (const float*)d_in[i++];
  const float* sa1_bn  = (const float*)d_in[i++];
  const float* sa2_wqk = (const float*)d_in[i++];
  const float* sa2_wv  = (const float*)d_in[i++];
  const float* sa2_bv  = (const float*)d_in[i++];
  const float* sa2_wt  = (const float*)d_in[i++];
  const float* sa2_bt  = (const float*)d_in[i++];
  const float* sa2_gn  = (const float*)d_in[i++];
  const float* sa2_bn  = (const float*)d_in[i++];
  i += 14;  // sa3, sa4 parameters: outputs x3, x4 are dead
  const float* fu_w = (const float*)d_in[i++];
  const float* fu_g = (const float*)d_in[i++];
  const float* fu_b = (const float*)d_in[i++];

  float* out1 = (float*)d_out;                 // [B,1024]
  float* outx = (float*)d_out + NBATCH * 1024; // [B,512,32]

  // ---- workspace layout ----
  float* W = (float*)d_ws;
  size_t off = 0;
  auto alloc = [&](size_t n) { float* p = W + off; off += n; return p; };
  float* stats = alloc(8192);     // stat arena (zeroed each launch)
  float* sbuf  = alloc(8192);     // per-channel scale/bias arena
  float* y2    = alloc((size_t)64 * BNCOLS);            // conv2 raw output
  float* nx1   = alloc((size_t)NBATCH * 64 * 3);
  float* nx2   = alloc((size_t)NBATCH * 32 * 3);
  float* ft1   = alloc((size_t)128 * NBATCH * 64 * 8);  // feat1 / reused as yl0b
  float* yl0a  = alloc((size_t)128 * NBATCH * 64 * 8);
  float* f0    = alloc((size_t)128 * NBATCH * 64);
  float* ft2   = alloc((size_t)256 * NBATCH * 32 * 8);  // feat2 / reused as yl1b
  float* yl1a  = alloc((size_t)256 * NBATCH * 32 * 8);
  float* xcat  = alloc((size_t)768 * NBATCH * 32);      // rows: x1 | x2 | f1
  float* zt    = alloc((size_t)256 * NBATCH * 32);
  float* emb   = alloc((size_t)256 * NBATCH * 32);
  float* hbuf  = alloc((size_t)256 * NBATCH * 32);
  float* qkb   = alloc((size_t)64  * NBATCH * 32);
  float* vb    = alloc((size_t)256 * NBATCH * 32);
  float* hxb   = alloc((size_t)256 * NBATCH * 32);
  float* tb    = alloc((size_t)256 * NBATCH * 32);
  float* yfu   = alloc((size_t)1024 * NBATCH * 32);
  float* f1    = xcat + (size_t)512 * NBATCH * 32;
  float* x1    = xcat;
  float* x2    = xcat + (size_t)256 * NBATCH * 32;

  // stat arena offsets
  float* stXM = stats + 0;     // 12
  float* stC2 = stats + 16;    // 128
  float* stL0A = stats + 160;  float* stL0B = stats + 416;
  float* stL1A = stats + 672;  float* stL1B = stats + 1184;
  float* stPT  = stats + 1696;
  float* stSA1 = stats + 2208; float* stSA2 = stats + 2720;
  float* stFU  = stats + 3232; // 2048
  // scale/bias offsets
  float* sbS1 = sbuf + 0;    float* sbS2 = sbuf + 128;
  float* sbL0A = sbuf + 256; float* sbL0B = sbuf + 512;
  float* sbL1A = sbuf + 768; float* sbL1B = sbuf + 1280;
  float* sbPT  = sbuf + 1792;
  float* sbSA1 = sbuf + 2304; float* sbSA2 = sbuf + 2816;
  float* sbFU  = sbuf + 3328; // 2048

  const int NC1 = NBATCH * 64 * 8;   // 32768
  const int NC2 = NBATCH * 32 * 8;   // 16384
  const int NCS = NBATCH * 32;       // 2048
  auto cdiv = [](int a, int b) { return (a + b - 1) / b; };

  // 0) zero stat arena
  pct_zero<<<cdiv(8192, 256), 256, 0, stream>>>(stats, 8192);
  // 1) x moments -> BN1 scale/bias (analytic: BN of a linear map of x)
  pct_xmom<<<2048, 256, 0, stream>>>(x, stXM);
  pct_bn1fin<<<1, 64, 0, stream>>>(stXM, w_c1, g1, b1, sbS1);
  // 2) fused conv1+bn1+relu+conv2 (WMMA) with BN2 stats
  pct_conv12<<<BNCOLS / 32, 256, 0, stream>>>(x, w_c1, sbS1, sbS1 + 64, w_c2, y2, stC2);
  pct_bnfin<<<1, 64, 0, stream>>>(stC2, g2, b2, sbS2, 64, (float)BNCOLS);
  // 3) group 1: knn + feature build (bn2+relu fused at gather)
  pct_group1<<<NBATCH * 64, 256, 0, stream>>>(x, idx1, y2, sbS2, sbS2 + 64, nx1, ft1);
  // 4) local_op 0 (128x128 WMMA GEMMs, BN fused)
  pct_gemm<<<dim3(NC1/32, 2), 256, 0, stream>>>(l0_w1, ft1, yl0a, nullptr, nullptr, 0,
                                                nullptr, stL0A, 128, 128, NC1);
  pct_bnfin<<<1, 128, 0, stream>>>(stL0A, l0_g1, l0_b1, sbL0A, 128, (float)NC1);
  pct_gemm<<<dim3(NC1/32, 2), 256, 0, stream>>>(l0_w2, yl0a, ft1, sbL0A, sbL0A + 128, 1,
                                                nullptr, stL0B, 128, 128, NC1);
  pct_bnfin<<<1, 128, 0, stream>>>(stL0B, l0_g2, l0_b2, sbL0B, 128, (float)NC1);
  pct_maxpool<<<cdiv(128 * NBATCH * 64, 256), 256, 0, stream>>>(ft1, sbL0B, sbL0B + 128,
                                                                f0, 128, NBATCH * 64);
  // 5) group 2 + local_op 1 (256x256 WMMA GEMMs)
  pct_group2<<<NBATCH * 32, 64, 0, stream>>>(nx1, idx2, f0, nx2, ft2);
  pct_gemm<<<dim3(NC2/32, 4), 256, 0, stream>>>(l1_w1, ft2, yl1a, nullptr, nullptr, 0,
                                                nullptr, stL1A, 256, 256, NC2);
  pct_bnfin<<<1, 256, 0, stream>>>(stL1A, l1_g1, l1_b1, sbL1A, 256, (float)NC2);
  pct_gemm<<<dim3(NC2/32, 4), 256, 0, stream>>>(l1_w2, yl1a, ft2, sbL1A, sbL1A + 256, 1,
                                                nullptr, stL1B, 256, 256, NC2);
  pct_bnfin<<<1, 256, 0, stream>>>(stL1B, l1_g2, l1_b2, sbL1B, 256, (float)NC2);
  pct_maxpool<<<cdiv(256 * NCS, 256), 256, 0, stream>>>(ft2, sbL1B, sbL1B + 256,
                                                        f1, 256, NCS);
  // 6) position embedding + pt conv
  pct_posemb<<<cdiv(256 * NCS, 256), 256, 0, stream>>>(nx2, w_pos, b_pos, emb);
  pct_gemm<<<dim3(NCS/32, 4), 256, 0, stream>>>(pt_w, f1, zt, nullptr, nullptr, 0,
                                                nullptr, stPT, 256, 256, NCS);
  pct_bnfin<<<1, 256, 0, stream>>>(stPT, pt_g, pt_b, sbPT, 256, (float)NCS);

  // 7) SA layers (only sa1, sa2 feed the output)
  auto run_sa = [&](const float* wqk, const float* wv, const float* bv,
                    const float* wt, const float* bt, const float* gn, const float* bnb,
                    const float* xin, const float* isc, const float* ibi,
                    float* stSA, float* sbSA, float* xout) {
    pct_addh<<<cdiv(256 * NCS, 256), 256, 0, stream>>>(xin, isc, ibi, emb, hbuf, NCS);
    pct_gemm<<<dim3(NCS/32, 1), 256, 0, stream>>>(wqk, hbuf, qkb, nullptr, nullptr, 0,
                                                  nullptr, nullptr, 64, 256, NCS);
    pct_gemm<<<dim3(NCS/32, 4), 256, 0, stream>>>(wv, hbuf, vb, nullptr, nullptr, 0,
                                                  bv, nullptr, 256, 256, NCS);
    pct_attn<<<NBATCH, 256, 0, stream>>>(qkb, vb, hbuf, hxb);
    pct_gemm<<<dim3(NCS/32, 4), 256, 0, stream>>>(wt, hxb, tb, nullptr, nullptr, 0,
                                                  bt, stSA, 256, 256, NCS);
    pct_bnfin<<<1, 256, 0, stream>>>(stSA, gn, bnb, sbSA, 256, (float)NCS);
    pct_saout<<<cdiv(256 * NCS, 256), 256, 0, stream>>>(hbuf, tb, sbSA, sbSA + 256, xout, NCS);
  };
  run_sa(sa1_wqk, sa1_wv, sa1_bv, sa1_wt, sa1_bt, sa1_gn, sa1_bn,
         zt, sbPT, sbPT + 256, stSA1, sbSA1, x1);       // z = relu(bn(zt)) fused
  run_sa(sa2_wqk, sa2_wv, sa2_bv, sa2_wt, sa2_bt, sa2_gn, sa2_bn,
         x1, nullptr, nullptr, stSA2, sbSA2, x2);

  // 8) fuse conv 1024x768 over cat=[x1;x2;f1] (contiguous rows of xcat)
  pct_gemm<<<dim3(NCS/32, 16), 256, 0, stream>>>(fu_w, xcat, yfu, nullptr, nullptr, 0,
                                                 nullptr, stFU, 1024, 768, NCS);
  pct_bnfin<<<cdiv(1024, 256), 256, 0, stream>>>(stFU, fu_g, fu_b, sbFU, 1024, (float)NCS);
  pct_fusemax<<<cdiv(NBATCH * 1024, 256), 256, 0, stream>>>(yfu, sbFU, sbFU + 1024, out1);
  pct_emitxpt<<<cdiv(NBATCH * 512 * 32, 256), 256, 0, stream>>>(xcat, outx);
}
